// CustomBertAttention_9818295238839
// MI455X (gfx1250) — matrix operand order repaired and checked by reading, not compiled, not run on hardware
//
#include <hip/hip_runtime.h>
#include <hip/hip_bf16.h>

// Problem dims
#define BB 4
#define SS 2048
#define HH 1024
#define NHH 16
#define HDD 64

typedef __attribute__((ext_vector_type(16))) __bf16 v16bf;
typedef __attribute__((ext_vector_type(8)))  float  v8f;

union FragAB { v16bf v; unsigned int u[8]; };

__device__ __forceinline__ unsigned short f2bf(float f) {
    unsigned int x = __float_as_uint(f);
    unsigned int r = x + 0x7FFFu + ((x >> 16) & 1u);
    return (unsigned short)(r >> 16);
}

__device__ __forceinline__ unsigned int pack_bf2(float lo, float hi) {
    return (unsigned int)f2bf(lo) | ((unsigned int)f2bf(hi) << 16);
}

// A fragment (16x32 bf16, MxK), src row-major [M][K], ld = row stride in elems.
// ISA layout: lane holds row M=lane%16; VGPR r holds K pair:
//   K = (r/4)*16 + (lane>=16 ? 8 : 0) + (r%4)*2 + {0,1}
__device__ __forceinline__ void load_frag_a(FragAB& fr, const unsigned short* src, int ld) {
    int lane = threadIdx.x & 31;
    int m = lane & 15, hi = lane >> 4;
#pragma unroll
    for (int r = 0; r < 8; ++r) {
        int k = ((r >> 2) << 4) + (hi << 3) + ((r & 3) << 1);
        fr.u[r] = *(const unsigned int*)(src + (size_t)m * ld + k);
    }
}

// Same A layout but source is fp32 in LDS (P tile); convert to bf16 pairs.
__device__ __forceinline__ void load_frag_a_f32(FragAB& fr, const float* src, int ld) {
    int lane = threadIdx.x & 31;
    int m = lane & 15, hi = lane >> 4;
#pragma unroll
    for (int r = 0; r < 8; ++r) {
        int k = ((r >> 2) << 4) + (hi << 3) + ((r & 3) << 1);
        fr.u[r] = pack_bf2(src[m * ld + k], src[m * ld + k + 1]);
    }
}

// B fragment (32x16 bf16, KxN) where memory holds src[n][k] row-major
// (weights W[n][k] for X@W.T, and K-matrix K[key][d] for Q@K.T).
// Lane n = lane%16; VGPR r holds K pair: K = (r/4)*16 + (lane>=16?8:0) + (r%4)*2 + {0,1}.
__device__ __forceinline__ void load_frag_b_nt(FragAB& fr, const unsigned short* src, int ld) {
    int lane = threadIdx.x & 31;
    int n = lane & 15, hi = lane >> 4;
#pragma unroll
    for (int r = 0; r < 8; ++r) {
        int k = ((r >> 2) << 4) + (hi << 3) + ((r & 3) << 1);
        fr.u[r] = *(const unsigned int*)(src + (size_t)n * ld + k);
    }
}

// B fragment (32x16) where memory holds src[k][n] row-major (V matrix: V[key][d]).
__device__ __forceinline__ void load_frag_b_nn(FragAB& fr, const unsigned short* src, int ld) {
    int lane = threadIdx.x & 31;
    int n = lane & 15, hi = lane >> 4;
#pragma unroll
    for (int r = 0; r < 8; ++r) {
        int k = ((r >> 2) << 4) + (hi << 3) + ((r & 3) << 1);
        unsigned int lo = src[(size_t)k * ld + n];
        unsigned int h2 = src[(size_t)(k + 1) * ld + n];
        fr.u[r] = lo | (h2 << 16);
    }
}

__device__ __forceinline__ v8f wmma_bf16(const FragAB& a, const FragAB& b, v8f c) {
    return __builtin_amdgcn_wmma_f32_16x16x32_bf16(false, a.v, false, b.v,
                                                   (short)0, c, false, false);
}

// ---------------- fp32 -> bf16 conversion ----------------
__global__ __launch_bounds__(256)
void cvt_bf16_kernel(const float* __restrict__ in, unsigned short* __restrict__ out, int n) {
    for (int i = blockIdx.x * blockDim.x + threadIdx.x; i < n; i += gridDim.x * blockDim.x)
        out[i] = f2bf(in[i]);
}

// ---------------- fused QKV projection ----------------
// One wave computes a 64x32 output tile: 4 A-frags x 2 B-frags -> 8 WMMAs/k-step.
__global__ __launch_bounds__(32)
void qkv_gemm_kernel(const unsigned short* __restrict__ X,
                     const unsigned short* __restrict__ W16,
                     const float* __restrict__ bq, const float* __restrict__ bk,
                     const float* __restrict__ bv,
                     unsigned short* __restrict__ Q, unsigned short* __restrict__ Kd,
                     unsigned short* __restrict__ V) {
    const int m0 = blockIdx.x * 64;
    const int n0 = blockIdx.y * 32;
    const int w  = blockIdx.z;  // 0=Q,1=K,2=V
    const unsigned short* Wsel = W16 + (size_t)w * HH * HH;
    const float* bias = (w == 0) ? bq : (w == 1) ? bk : bv;
    unsigned short* Out = (w == 0) ? Q : (w == 1) ? Kd : V;

    v8f acc[4][2];
#pragma unroll
    for (int i = 0; i < 4; ++i)
#pragma unroll
        for (int j2 = 0; j2 < 2; ++j2)
#pragma unroll
            for (int j = 0; j < 8; ++j) acc[i][j2][j] = 0.0f;

    for (int k0 = 0; k0 < HH; k0 += 32) {
        FragAB a[4], b[2];
#pragma unroll
        for (int i = 0; i < 4; ++i)
            load_frag_a(a[i], X + (size_t)(m0 + 16 * i) * HH + k0, HH);
#pragma unroll
        for (int j2 = 0; j2 < 2; ++j2)
            load_frag_b_nt(b[j2], Wsel + (size_t)(n0 + 16 * j2) * HH + k0, HH);
#pragma unroll
        for (int i = 0; i < 4; ++i)
#pragma unroll
            for (int j2 = 0; j2 < 2; ++j2)
                acc[i][j2] = wmma_bf16(a[i], b[j2], acc[i][j2]);
    }

    const int lane = threadIdx.x & 31, hi = lane >> 4, nc = lane & 15;
#pragma unroll
    for (int i = 0; i < 4; ++i)
#pragma unroll
        for (int j2 = 0; j2 < 2; ++j2) {
            int n = n0 + 16 * j2 + nc;
            int h = n >> 6, d = n & 63;
            float bval = bias[n];
#pragma unroll
            for (int j = 0; j < 8; ++j) {
                int m = m0 + 16 * i + j + hi * 8;
                int bidx = m >> 11, s = m & 2047;
                size_t idx = (((size_t)bidx * NHH + h) * SS + s) * HDD + d;
                Out[idx] = f2bf(acc[i][j2][j] + bval);
            }
        }
}

// ---------------- flash attention ----------------
// One wave per (32 query rows, head). K/V fragments reused across both q-tiles.
__global__ __launch_bounds__(32)
void attention_kernel(const unsigned short* __restrict__ Q,
                      const unsigned short* __restrict__ Kd,
                      const unsigned short* __restrict__ V,
                      const float* __restrict__ biasM,
                      const float* __restrict__ biasCoef,
                      unsigned short* __restrict__ CTX) {
    __shared__ float pTile[2 * 16 * 32];

    const int q0 = blockIdx.x * 32;
    const int bh = blockIdx.y;
    const float coef = biasCoef[0];
    const float scl = 0.125f;  // 1/sqrt(HD=64)

    const unsigned short* Qp = Q  + ((size_t)bh * SS + q0) * HDD;
    const unsigned short* Kp = Kd + (size_t)bh * SS * HDD;
    const unsigned short* Vp = V  + (size_t)bh * SS * HDD;

    FragAB aQ[2][2];
#pragma unroll
    for (int qi = 0; qi < 2; ++qi) {
        load_frag_a(aQ[qi][0], Qp + (size_t)(16 * qi) * HDD, HDD);       // d 0..31
        load_frag_a(aQ[qi][1], Qp + (size_t)(16 * qi) * HDD + 32, HDD);  // d 32..63
    }

    const int lane = threadIdx.x & 31, hi = lane >> 4, nc = lane & 15;

    float rowmax[2][8], rowsum[2][8];
    v8f o[2][4];
#pragma unroll
    for (int qi = 0; qi < 2; ++qi) {
#pragma unroll
        for (int j = 0; j < 8; ++j) { rowmax[qi][j] = -1e30f; rowsum[qi][j] = 0.0f; }
#pragma unroll
        for (int t = 0; t < 4; ++t)
#pragma unroll
            for (int j = 0; j < 8; ++j) o[qi][t][j] = 0.0f;
    }

    for (int kv = 0; kv < SS; kv += 32) {
        // prefetch next 32-key strip of K and V (uniform guard keeps EXEC full)
        if (kv + 32 < SS) {
            __builtin_prefetch(Kp + (size_t)(kv + 32) * HDD + (size_t)lane * 64, 0, 3);
            __builtin_prefetch(Vp + (size_t)(kv + 32) * HDD + (size_t)lane * 64, 0, 3);
        }

        // K fragments (shared across both q-tiles)
        FragAB bK[2][2];  // [key-tile][d-half]
#pragma unroll
        for (int kt = 0; kt < 2; ++kt) {
            load_frag_b_nt(bK[kt][0], Kp + (size_t)(kv + kt * 16) * HDD, HDD);
            load_frag_b_nt(bK[kt][1], Kp + (size_t)(kv + kt * 16) * HDD + 32, HDD);
        }

        v8f sc[2][2];
#pragma unroll
        for (int qi = 0; qi < 2; ++qi)
#pragma unroll
            for (int kt = 0; kt < 2; ++kt) {
                v8f c;
#pragma unroll
                for (int j = 0; j < 8; ++j) c[j] = 0.0f;
                c = wmma_bf16(aQ[qi][0], bK[kt][0], c);
                c = wmma_bf16(aQ[qi][1], bK[kt][1], c);
                sc[qi][kt] = c;
            }

        // scale + additive bias
#pragma unroll
        for (int qi = 0; qi < 2; ++qi)
#pragma unroll
            for (int kt = 0; kt < 2; ++kt)
#pragma unroll
                for (int j = 0; j < 8; ++j) {
                    int row = q0 + 16 * qi + j + hi * 8;
                    int col = kv + 16 * kt + nc;
                    sc[qi][kt][j] = sc[qi][kt][j] * scl +
                                    biasM[(size_t)row * SS + col] * coef;
                }

        // online softmax per q-tile; row lives in (j, half), 16 cols across lanes
#pragma unroll
        for (int qi = 0; qi < 2; ++qi)
#pragma unroll
            for (int j = 0; j < 8; ++j) {
                float mx = fmaxf(sc[qi][0][j], sc[qi][1][j]);
                for (int m = 8; m >= 1; m >>= 1) mx = fmaxf(mx, __shfl_xor(mx, m, 32));
                float nm = fmaxf(rowmax[qi][j], mx);
                float resc = __expf(rowmax[qi][j] - nm);
                rowmax[qi][j] = nm;
                float p0 = __expf(sc[qi][0][j] - nm);
                float p1 = __expf(sc[qi][1][j] - nm);
                float ps = p0 + p1;
                for (int m = 8; m >= 1; m >>= 1) ps += __shfl_xor(ps, m, 32);
                rowsum[qi][j] = rowsum[qi][j] * resc + ps;
#pragma unroll
                for (int t = 0; t < 4; ++t) o[qi][t][j] *= resc;
                int rl = j + hi * 8;
                pTile[qi * 512 + rl * 32 + nc]      = p0;
                pTile[qi * 512 + rl * 32 + 16 + nc] = p1;
            }
        __syncthreads();

        // V fragments (shared across both q-tiles)
        FragAB bV[4];
#pragma unroll
        for (int t = 0; t < 4; ++t)
            load_frag_b_nn(bV[t], Vp + (size_t)kv * HDD + t * 16, HDD);

#pragma unroll
        for (int qi = 0; qi < 2; ++qi) {
            FragAB aP;
            load_frag_a_f32(aP, pTile + qi * 512, 32);
#pragma unroll
            for (int t = 0; t < 4; ++t)
                o[qi][t] = wmma_bf16(aP, bV[t], o[qi][t]);
        }
        __syncthreads();
    }

    // epilogue: ctx -> [B, S, H] bf16
    const int b = bh >> 4, h = bh & 15;
#pragma unroll
    for (int qi = 0; qi < 2; ++qi)
#pragma unroll
        for (int t = 0; t < 4; ++t)
#pragma unroll
            for (int j = 0; j < 8; ++j) {
                int s = q0 + 16 * qi + j + hi * 8;
                float val = o[qi][t][j] / rowsum[qi][j];
                size_t idx = ((size_t)b * SS + s) * HH + h * 64 + t * 16 + nc;
                CTX[idx] = f2bf(val);
            }
}

// ---------------- output projection + residual ----------------
// One wave computes a 64x32 output tile.
__global__ __launch_bounds__(32)
void out_gemm_kernel(const unsigned short* __restrict__ CTX,
                     const unsigned short* __restrict__ Wo16,
                     const float* __restrict__ bo,
                     const float* __restrict__ hidden,
                     float* __restrict__ Xout) {
    const int m0 = blockIdx.x * 64;
    const int n0 = blockIdx.y * 32;

    v8f acc[4][2];
#pragma unroll
    for (int i = 0; i < 4; ++i)
#pragma unroll
        for (int j2 = 0; j2 < 2; ++j2)
#pragma unroll
            for (int j = 0; j < 8; ++j) acc[i][j2][j] = 0.0f;

    for (int k0 = 0; k0 < HH; k0 += 32) {
        FragAB a[4], b[2];
#pragma unroll
        for (int i = 0; i < 4; ++i)
            load_frag_a(a[i], CTX + (size_t)(m0 + 16 * i) * HH + k0, HH);
#pragma unroll
        for (int j2 = 0; j2 < 2; ++j2)
            load_frag_b_nt(b[j2], Wo16 + (size_t)(n0 + 16 * j2) * HH + k0, HH);
#pragma unroll
        for (int i = 0; i < 4; ++i)
#pragma unroll
            for (int j2 = 0; j2 < 2; ++j2)
                acc[i][j2] = wmma_bf16(a[i], b[j2], acc[i][j2]);
    }

    const int lane = threadIdx.x & 31, hi = lane >> 4, nc = lane & 15;
#pragma unroll
    for (int i = 0; i < 4; ++i)
#pragma unroll
        for (int j2 = 0; j2 < 2; ++j2) {
            int n = n0 + 16 * j2 + nc;
            float bval = bo[n];
#pragma unroll
            for (int j = 0; j < 8; ++j) {
                int m = m0 + 16 * i + j + hi * 8;
                size_t idx = (size_t)m * HH + n;
                Xout[idx] = acc[i][j2][j] + bval + hidden[idx];
            }
        }
}

// ---------------- layernorm ----------------
__device__ __forceinline__ float block_reduce_sum(float v, float* red) {
    for (int m = 16; m >= 1; m >>= 1) v += __shfl_xor(v, m, 32);
    int wid = threadIdx.x >> 5, lane = threadIdx.x & 31;
    if (lane == 0) red[wid] = v;
    __syncthreads();
    float t = (threadIdx.x < 8) ? red[threadIdx.x] : 0.0f;
    if (threadIdx.x < 32)
        for (int m = 4; m >= 1; m >>= 1) t += __shfl_xor(t, m, 32);
    if (threadIdx.x == 0) red[0] = t;
    __syncthreads();
    float r = red[0];
    __syncthreads();
    return r;
}

__global__ __launch_bounds__(256)
void layernorm_kernel(const float* __restrict__ Xin,
                      const float* __restrict__ gamma,
                      const float* __restrict__ beta,
                      float* __restrict__ out) {
    __shared__ float red[8];
    const size_t row = blockIdx.x;
    const float* x = Xin + row * HH;

    float s = 0.0f;
    for (int i = threadIdx.x; i < HH; i += 256) s += x[i];
    float mu = block_reduce_sum(s, red) * (1.0f / HH);

    float v = 0.0f;
    for (int i = threadIdx.x; i < HH; i += 256) {
        float d = x[i] - mu;
        v += d * d;
    }
    float var = block_reduce_sum(v, red) * (1.0f / HH);
    float inv = rsqrtf(var + 1e-12f);

    float* o = out + row * HH;
    for (int i = threadIdx.x; i < HH; i += 256)
        o[i] = (x[i] - mu) * inv * gamma[i] + beta[i];
}

// ---------------- launch ----------------
extern "C" void kernel_launch(void* const* d_in, const int* in_sizes, int n_in,
                              void* d_out, int out_size, void* d_ws, size_t ws_size,
                              hipStream_t stream) {
    const float* hidden   = (const float*)d_in[0];   // [4,2048,1024]
    const float* biasMat  = (const float*)d_in[1];   // [2048,2048]
    const float* biasCoef = (const float*)d_in[2];   // scalar
    const float* Wq = (const float*)d_in[3];
    const float* bq = (const float*)d_in[4];
    const float* Wk = (const float*)d_in[5];
    const float* bk = (const float*)d_in[6];
    const float* Wv = (const float*)d_in[7];
    const float* bv = (const float*)d_in[8];
    const float* Wo = (const float*)d_in[9];
    const float* bo = (const float*)d_in[10];
    const float* gamma = (const float*)d_in[11];
    const float* beta  = (const float*)d_in[12];
    float* out = (float*)d_out;

    const size_t nTok = (size_t)BB * SS;   // 8192
    const size_t nHid = nTok * HH;         // 8388608
    const size_t nW   = (size_t)HH * HH;   // 1048576

    // workspace layout
    char* ws = (char*)d_ws;
    unsigned short* X16 = (unsigned short*)(ws);                      // 16 MB
    unsigned short* W16 = (unsigned short*)(ws + nHid * 2);           //  8 MB (Wq,Wk,Wv,Wo)
    unsigned short* Q16 = (unsigned short*)(ws + nHid * 2 + nW * 8);  // 16 MB
    unsigned short* K16 = Q16 + nHid;                                 // 16 MB
    unsigned short* V16 = K16 + nHid;                                 // 16 MB
    unsigned short* CTX = V16 + nHid;                                 // 16 MB
    float*          XO  = (float*)(CTX + nHid);                       // 32 MB

    // 1. convert hidden + weights to bf16
    cvt_bf16_kernel<<<dim3(4096), dim3(256), 0, stream>>>(hidden, X16, (int)nHid);
    cvt_bf16_kernel<<<dim3(2048), dim3(256), 0, stream>>>(Wq, W16 + 0 * nW, (int)nW);
    cvt_bf16_kernel<<<dim3(2048), dim3(256), 0, stream>>>(Wk, W16 + 1 * nW, (int)nW);
    cvt_bf16_kernel<<<dim3(2048), dim3(256), 0, stream>>>(Wv, W16 + 2 * nW, (int)nW);
    cvt_bf16_kernel<<<dim3(2048), dim3(256), 0, stream>>>(Wo, W16 + 3 * nW, (int)nW);

    // 2. QKV projections (one wave per 64x32 output tile)
    qkv_gemm_kernel<<<dim3(nTok / 64, HH / 32, 3), dim3(32), 0, stream>>>(
        X16, W16, bq, bk, bv, Q16, K16, V16);

    // 3. flash attention (one wave per 32 query rows x head)
    attention_kernel<<<dim3(SS / 32, BB * NHH), dim3(32), 0, stream>>>(
        Q16, K16, V16, biasMat, biasCoef, CTX);

    // 4. output projection + residual
    out_gemm_kernel<<<dim3(nTok / 64, HH / 32), dim3(32), 0, stream>>>(
        CTX, W16 + 3 * nW, bo, hidden, XO);

    // 5. layernorm
    layernorm_kernel<<<dim3(nTok), dim3(256), 0, stream>>>(XO, gamma, beta, out);
}
